// Graphormer3DGraphEncoder_67250597921317
// MI455X (gfx1250) — compile-verified, matrix operand order checked by reading, not run on hardware
//
#include <hip/hip_runtime.h>

typedef _Float16 f16;
typedef __attribute__((ext_vector_type(16))) _Float16 v16h;
typedef __attribute__((ext_vector_type(8)))  float    v8f;
typedef int i32x4 __attribute__((vector_size(16)));

#define DEV __device__ __forceinline__

#if defined(__has_builtin)
#if __has_builtin(__builtin_amdgcn_global_load_async_to_lds_b128) && \
    __has_builtin(__builtin_amdgcn_s_wait_asynccnt)
#define USE_ASYNC_LDS 1
#endif
#endif
#ifndef USE_ASYNC_LDS
#define USE_ASYNC_LDS 0
#endif

namespace {

constexpr int Bx = 16, Nn = 64, Sx = 65, Hh = 768, HEADS = 12, HD = 64;
constexpr int FFN = 3072, GK = 128, LAYERS = 4;
constexpr int ROWS  = Sx * Bx;        // 1040 tokens (S,B) flattened as s*16+b
constexpr int MBIAS = Bx * Sx * Sx;   // 67600 bias rows
constexpr float SCALE = 0.125f;       // HD^-0.5

DEV float gelu_exact(float x) { return 0.5f * x * (1.0f + erff(x * 0.70710678118f)); }

DEV void async_load_b128(const void* g, void* l) {
#if USE_ASYNC_LDS
  __builtin_amdgcn_global_load_async_to_lds_b128(
      (__attribute__((address_space(1))) i32x4*)g,
      (__attribute__((address_space(3))) i32x4*)l, 0, 0);
#else
  (void)g; (void)l;
#endif
}

// ---------------------------------------------------------------- f32 -> f16
__global__ void k_cvt(const float* __restrict__ src, f16* __restrict__ dst, int n) {
  int i = blockIdx.x * blockDim.x + threadIdx.x;
  int stride = gridDim.x * blockDim.x;
  for (; i < n; i += stride) dst[i] = (f16)src[i];
}

// ------------------------------------------------- atom gathers + rank-1 dots
__global__ __launch_bounds__(256) void k_embed(
    const int* __restrict__ nodes, const float* __restrict__ ein,
    const float* __restrict__ eout, const float* __restrict__ gmw,
    const float* __restrict__ gbw, float* __restrict__ input_emb,
    float* __restrict__ out_emb, float* __restrict__ em, int* __restrict__ pmask) {
  int bn = blockIdx.x;                       // b*N + n
  int b = bn / Nn, n = bn - b * Nn;
  const int* idx = nodes + bn * 9;
  int id[9];
#pragma unroll
  for (int f = 0; f < 9; ++f) id[f] = idx[f];
  float d0 = 0.f, d1 = 0.f, d2 = 0.f, d3 = 0.f;
  for (int h = threadIdx.x; h < Hh; h += 256) {
    float si = 0.f, so = 0.f;
#pragma unroll
    for (int f = 0; f < 9; ++f) {
      si += ein[id[f] * Hh + h];
      so += eout[id[f] * Hh + h];
    }
    input_emb[(size_t)(b * Sx + n + 1) * Hh + h] = si;
    out_emb[(size_t)bn * Hh + h] = so;
    float wm = gmw[h], wb = gbw[h];
    d0 += si * wm; d1 += si * wb; d2 += so * wm; d3 += so * wb;
  }
  __shared__ float r0[256], r1[256], r2[256], r3[256];
  int t = threadIdx.x;
  r0[t] = d0; r1[t] = d1; r2[t] = d2; r3[t] = d3;
  __syncthreads();
  for (int off = 128; off > 0; off >>= 1) {
    if (t < off) { r0[t] += r0[t+off]; r1[t] += r1[t+off]; r2[t] += r2[t+off]; r3[t] += r3[t+off]; }
    __syncthreads();
  }
  if (t == 0) {
    em[0 * Bx * Nn + bn] = r0[0];
    em[1 * Bx * Nn + bn] = r1[0];
    em[2 * Bx * Nn + bn] = r2[0];
    em[3 * Bx * Nn + bn] = r3[0];
    pmask[b * Sx + n + 1] = (id[0] == 0) ? 1 : 0;
    if (n == 0) pmask[b * Sx] = 0;
  }
}

__global__ void k_token(const float* __restrict__ gt, float* __restrict__ input_emb) {
  int b = blockIdx.x;
  for (int h = threadIdx.x; h < Hh; h += blockDim.x)
    input_emb[(size_t)b * Sx * Hh + h] = gt[h];
}

// ------------------------------------------------------------- layernorms
__global__ __launch_bounds__(256) void k_ln(const float* __restrict__ x,
    const float* __restrict__ g, const float* __restrict__ bb,
    float* __restrict__ outf, f16* __restrict__ outh) {
  __shared__ float rs[256], rq[256];
  size_t row = blockIdx.x;
  const float* xr = x + row * Hh;
  float s = 0.f, q = 0.f;
  for (int h = threadIdx.x; h < Hh; h += 256) { float v = xr[h]; s += v; q += v * v; }
  int t = threadIdx.x;
  rs[t] = s; rq[t] = q; __syncthreads();
  for (int off = 128; off > 0; off >>= 1) {
    if (t < off) { rs[t] += rs[t+off]; rq[t] += rq[t+off]; }
    __syncthreads();
  }
  float mu = rs[0] / Hh;
  float var = rq[0] / Hh - mu * mu;
  float inv = rsqrtf(var + 1e-5f);
  for (int h = threadIdx.x; h < Hh; h += 256) {
    float v = (xr[h] - mu) * inv * g[h] + bb[h];
    outf[row * Hh + h] = v;
    outh[row * Hh + h] = (f16)v;
  }
}

// LN of input_emb [B,S,H] -> token stream [s*16+b, H]
__global__ __launch_bounds__(256) void k_ln_emb(const float* __restrict__ input_emb,
    const float* __restrict__ g, const float* __restrict__ bb,
    float* __restrict__ outf, f16* __restrict__ outh) {
  __shared__ float rs[256], rq[256];
  int r = blockIdx.x;           // s*16+b
  int s = r >> 4, b = r & 15;
  const float* xr = input_emb + (size_t)(b * Sx + s) * Hh;
  float sm = 0.f, q = 0.f;
  for (int h = threadIdx.x; h < Hh; h += 256) { float v = xr[h]; sm += v; q += v * v; }
  int t = threadIdx.x;
  rs[t] = sm; rq[t] = q; __syncthreads();
  for (int off = 128; off > 0; off >>= 1) {
    if (t < off) { rs[t] += rs[t+off]; rq[t] += rq[t+off]; }
    __syncthreads();
  }
  float mu = rs[0] / Hh;
  float var = rq[0] / Hh - mu * mu;
  float inv = rsqrtf(var + 1e-5f);
  for (int h = threadIdx.x; h < Hh; h += 256) {
    float v = (xr[h] - mu) * inv * g[h] + bb[h];
    outf[(size_t)r * Hh + h] = v;
    outh[(size_t)r * Hh + h] = (f16)v;
  }
}

// ----------------------------------------- Gaussian basis features (f16 A op)
__global__ void k_gbf(const float* __restrict__ pos, const float* __restrict__ em,
                      const float* __restrict__ means, const float* __restrict__ stds,
                      const float* __restrict__ mul_b, const float* __restrict__ bias_b,
                      const float* __restrict__ clf, f16* __restrict__ gbf) {
  int r = blockIdx.x;                 // b*S*S + i*S + j
  int k = threadIdx.x;                // 0..127
  int b = r / (Sx * Sx);
  int rem = r - b * Sx * Sx;
  int i = rem / Sx, j = rem - (rem / Sx) * Sx;
  float g;
  if (i == 0 || j == 0) {
    g = clf[k];
  } else {
    int ni = i - 1, nj = j - 1;
    const float* pi = pos + (size_t)(b * Nn + ni) * 3;
    const float* pj = pos + (size_t)(b * Nn + nj) * 3;
    float dx = pi[0] - pj[0], dy = pi[1] - pj[1], dz = pi[2] - pj[2];
    float dist = sqrtf(dx * dx + dy * dy + dz * dz + 1e-12f);
    float mul = em[0 * Bx * Nn + b * Nn + ni] + em[2 * Bx * Nn + b * Nn + nj] + mul_b[0];
    float add = em[1 * Bx * Nn + b * Nn + ni] + em[3 * Bx * Nn + b * Nn + nj] + bias_b[0];
    float x = mul * dist + add;
    float sd = fabsf(stds[k]) + 1e-5f;
    float tt = (x - means[k]) / sd;
    g = expf(-0.5f * tt * tt) / (2.5066282746310002f * sd);
  }
  gbf[(size_t)r * GK + k] = (f16)g;
}

// ------------------------------------------------ WMMA GEMM (64x128 block tile)
// 8 waves, each computing a 32x32 output tile (4 WMMAs / K-step).
// A staged row-major (padded), B staged *transposed* so both fragment types are
// two 16B ds_load_b128 per lane (per the ISA 16-bit operand layout).
enum { EPI_GELU_F16 = 0, EPI_F32_RESID = 1, EPI_QKV = 2 };

struct GemmP {
  const f16* A; const f16* Bw; const float* bias;
  const float* resid; float* outf; f16* outh;
  f16* q; f16* k; f16* v;
  int M, N, Kd;
};

template <int EPI>
__global__ __launch_bounds__(256) void k_gemm(GemmP p) {
  __shared__ __align__(16) f16 As[64][40];    // 64 x 32 (+8 pad)
  __shared__ __align__(16) f16 Bt[128][40];   // transposed: Bt[n][k], 128 x 32 (+8)
  const int tid  = threadIdx.x;
  const int lane = tid & 31;
  const int w    = tid >> 5;                  // 8 waves: 2 in M x 4 in N
  const int mw   = (w & 1) << 5;              // 0 / 32
  const int nw   = (w >> 1) << 5;             // 0 / 32 / 64 / 96
  const int lh   = lane & 15;
  const int kb   = (lane < 16) ? 0 : 8;
  const int m0   = blockIdx.y * 64;
  const int n0   = blockIdx.x * 128;
  const int ar = tid >> 2, ac = (tid & 3) << 3;   // A stage: 64 rows x 4x(8 halves)
  const int kr = tid >> 3, nb = (tid & 7) << 4;   // B stage: 32 k-rows x 8x(16 n)

  union V16 { v16h v; uint4 q[2]; f16 e[16]; };
  union V8F { v8f v; float e[8]; };
  union H8  { uint4 q; f16 e[8]; };
  V8F acc[2][2];
#pragma unroll
  for (int ti = 0; ti < 2; ++ti)
#pragma unroll
    for (int tj = 0; tj < 2; ++tj)
#pragma unroll
      for (int r = 0; r < 8; ++r) acc[ti][tj].e[r] = 0.f;

  for (int k0 = 0; k0 < p.Kd; k0 += 32) {
    const f16* gA = p.A + (size_t)(m0 + ar) * p.Kd + k0 + ac;
#if !USE_ASYNC_LDS
    uint4 av = make_uint4(0u, 0u, 0u, 0u);
    if (m0 + ar < p.M) av = *reinterpret_cast<const uint4*>(gA);
#endif
    H8 hb0, hb1;
    hb0.q = *reinterpret_cast<const uint4*>(p.Bw + (size_t)(k0 + kr) * p.N + n0 + nb);
    hb1.q = *reinterpret_cast<const uint4*>(p.Bw + (size_t)(k0 + kr) * p.N + n0 + nb + 8);
    if (k0 + 32 < p.Kd) {   // hint next K tile into cache (global_prefetch_b8)
      __builtin_prefetch(gA + 32, 0, 3);
      __builtin_prefetch(p.Bw + (size_t)(k0 + 32 + kr) * p.N + n0 + nb, 0, 3);
    }
    __syncthreads();        // previous iteration's fragment reads complete
#if USE_ASYNC_LDS
    if (m0 + ar < p.M) async_load_b128(gA, &As[ar][ac]);
#else
    *reinterpret_cast<uint4*>(&As[ar][ac]) = av;
#endif
#pragma unroll
    for (int e = 0; e < 8; ++e) {           // transpose B into Bt[n][k]
      Bt[nb + e][kr]     = hb0.e[e];
      Bt[nb + 8 + e][kr] = hb1.e[e];
    }
#if USE_ASYNC_LDS
    __builtin_amdgcn_s_wait_asynccnt(0);
#endif
    __syncthreads();

    const f16* ar0 = &As[mw + lh][0];
    const f16* ar1 = &As[mw + 16 + lh][0];
    const f16* br0 = &Bt[nw + lh][0];
    const f16* br1 = &Bt[nw + 16 + lh][0];
    V16 a0, a1, b0, b1;
    a0.q[0] = *reinterpret_cast<const uint4*>(ar0 + kb);
    a0.q[1] = *reinterpret_cast<const uint4*>(ar0 + 16 + kb);
    a1.q[0] = *reinterpret_cast<const uint4*>(ar1 + kb);
    a1.q[1] = *reinterpret_cast<const uint4*>(ar1 + 16 + kb);
    b0.q[0] = *reinterpret_cast<const uint4*>(br0 + kb);
    b0.q[1] = *reinterpret_cast<const uint4*>(br0 + 16 + kb);
    b1.q[0] = *reinterpret_cast<const uint4*>(br1 + kb);
    b1.q[1] = *reinterpret_cast<const uint4*>(br1 + 16 + kb);
    acc[0][0].v = __builtin_amdgcn_wmma_f32_16x16x32_f16(false, a0.v, false, b0.v, (short)0, acc[0][0].v, false, false);
    acc[0][1].v = __builtin_amdgcn_wmma_f32_16x16x32_f16(false, a0.v, false, b1.v, (short)0, acc[0][1].v, false, false);
    acc[1][0].v = __builtin_amdgcn_wmma_f32_16x16x32_f16(false, a1.v, false, b0.v, (short)0, acc[1][0].v, false, false);
    acc[1][1].v = __builtin_amdgcn_wmma_f32_16x16x32_f16(false, a1.v, false, b1.v, (short)0, acc[1][1].v, false, false);
  }

#pragma unroll
  for (int ti = 0; ti < 2; ++ti) {
#pragma unroll
    for (int r = 0; r < 8; ++r) {
      int row = m0 + mw + ti * 16 + ((lane < 16) ? r : r + 8);
      if (row >= p.M) continue;
#pragma unroll
      for (int tj = 0; tj < 2; ++tj) {
        int col = n0 + nw + tj * 16 + lh;
        float val = acc[ti][tj].e[r] + p.bias[col];
        if constexpr (EPI == EPI_GELU_F16) {
          p.outh[(size_t)row * p.N + col] = (f16)gelu_exact(val);
        } else if constexpr (EPI == EPI_F32_RESID) {
          p.outf[(size_t)row * p.N + col] = val + p.resid[(size_t)row * p.N + col];
        } else { // EPI_QKV: scatter into q/k/v [B*HEADS, S, HD] f16, q scaled
          int which = col / Hh;
          int h = col - which * Hh;
          int head = h >> 6, d = h & 63;
          int s = row >> 4, bidx = row & 15;
          int z = bidx * HEADS + head;
          if (which == 0) val *= SCALE;
          f16* dst = (which == 0) ? p.q : (which == 1) ? p.k : p.v;
          dst[(size_t)(z * Sx + s) * HD + d] = (f16)val;
        }
      }
    }
  }
}

// -------------------------- 768 -> HEADS projection + mask -> attn_bias layout
__global__ __launch_bounds__(256) void k_bias2(const f16* __restrict__ C1,
    const f16* __restrict__ w2, const float* __restrict__ b2,
    const int* __restrict__ pmask, float* __restrict__ attn_bias) {
  int wave = threadIdx.x >> 5;
  int lane = threadIdx.x & 31;
  int r = blockIdx.x * 8 + wave;                 // one row per wave
  float acc[HEADS];
#pragma unroll
  for (int h = 0; h < HEADS; ++h) acc[h] = 0.f;
  const f16* arow = C1 + (size_t)r * Hh;
  for (int c = lane; c < Hh; c += 32) {
    float a = (float)arow[c];
    const f16* wr = w2 + c * HEADS;
#pragma unroll
    for (int h = 0; h < HEADS; ++h) acc[h] += a * (float)wr[h];
  }
#pragma unroll
  for (int h = 0; h < HEADS; ++h)
    for (int off = 16; off > 0; off >>= 1)
      acc[h] += __shfl_down(acc[h], off);
  if (lane == 0) {
    int b = r / (Sx * Sx);
    int rem = r - b * Sx * Sx;
    int i = rem / Sx, j = rem - (rem / Sx) * Sx;
    bool masked = pmask[b * Sx + j] != 0;
#pragma unroll
    for (int h = 0; h < HEADS; ++h) {
      float v = masked ? -__builtin_inff() : (acc[h] + b2[h]);
      attn_bias[((size_t)(b * HEADS + h) * Sx + i) * Sx + j] = v;
    }
  }
}

// ------------------------------------------- attention (S=65, per (b,head))
__global__ __launch_bounds__(256) void k_attn(const f16* __restrict__ q,
    const f16* __restrict__ kk, const f16* __restrict__ v,
    const float* __restrict__ attn_bias, f16* __restrict__ outh) {
  __shared__ f16 qs[Sx * HD], ks[Sx * HD], vs[Sx * HD];
  __shared__ float sc[Sx][Sx + 1];
  int z = blockIdx.x;
  int b = z / HEADS, head = z - b * HEADS;
  for (int t = threadIdx.x; t < Sx * HD; t += 256) {
    qs[t] = q[(size_t)z * Sx * HD + t];
    ks[t] = kk[(size_t)z * Sx * HD + t];
    vs[t] = v[(size_t)z * Sx * HD + t];
  }
  __syncthreads();
  for (int pix = threadIdx.x; pix < Sx * Sx; pix += 256) {
    int i = pix / Sx, j = pix - (pix / Sx) * Sx;
    float s = 0.f;
#pragma unroll 8
    for (int d = 0; d < HD; ++d) s += (float)qs[i * HD + d] * (float)ks[j * HD + d];
    sc[i][j] = s + attn_bias[(size_t)z * Sx * Sx + pix];
  }
  __syncthreads();
  if (threadIdx.x < Sx) {
    int i = threadIdx.x;
    float mx = -__builtin_inff();
    for (int j = 0; j < Sx; ++j) mx = fmaxf(mx, sc[i][j]);
    float sum = 0.f;
    for (int j = 0; j < Sx; ++j) { float e = expf(sc[i][j] - mx); sc[i][j] = e; sum += e; }
    float inv = 1.f / sum;
    for (int j = 0; j < Sx; ++j) sc[i][j] *= inv;
  }
  __syncthreads();
  for (int o = threadIdx.x; o < Sx * HD; o += 256) {
    int i = o >> 6, d = o & 63;
    float s = 0.f;
    for (int j = 0; j < Sx; ++j) s += sc[i][j] * (float)vs[j * HD + d];
    outh[(size_t)(i * Bx + b) * Hh + head * HD + d] = (f16)s;   // [s*16+b, H]
  }
}

} // namespace

extern "C" void kernel_launch(void* const* d_in, const int* in_sizes, int n_in,
                              void* d_out, int out_size, void* d_ws, size_t ws_size,
                              hipStream_t stream) {
  const int*   input_nodes = (const int*)  d_in[0];
  const float* pos         = (const float*)d_in[1];
  const float* atom_in     = (const float*)d_in[2];
  const float* graph_tok   = (const float*)d_in[3];
  const float* atom_out    = (const float*)d_in[4];
  const float* gbf_means   = (const float*)d_in[5];
  const float* gbf_stds    = (const float*)d_in[6];
  const float* gbf_mul_w   = (const float*)d_in[7];
  const float* gbf_mul_b   = (const float*)d_in[8];
  const float* gbf_bias_w  = (const float*)d_in[9];
  const float* gbf_bias_b  = (const float*)d_in[10];
  const float* clf_gbf     = (const float*)d_in[11];
  const float* bp_w1       = (const float*)d_in[12];
  const float* bp_b1       = (const float*)d_in[13];
  const float* bp_w2       = (const float*)d_in[14];
  const float* bp_b2       = (const float*)d_in[15];
  const float* emb_ln_g    = (const float*)d_in[16];
  const float* emb_ln_b    = (const float*)d_in[17];
  const float* w_qkv       = (const float*)d_in[18];
  const float* b_qkv       = (const float*)d_in[19];
  const float* w_o         = (const float*)d_in[20];
  const float* b_o         = (const float*)d_in[21];
  const float* ln1_g       = (const float*)d_in[22];
  const float* ln1_b       = (const float*)d_in[23];
  const float* w1          = (const float*)d_in[24];
  const float* b1          = (const float*)d_in[25];
  const float* w2          = (const float*)d_in[26];
  const float* b2          = (const float*)d_in[27];
  const float* ln2_g       = (const float*)d_in[28];
  const float* ln2_b       = (const float*)d_in[29];

  char* p = (char*)d_ws;
  auto carve = [&p](size_t bytes) { char* r = p; p += (bytes + 255) & ~(size_t)255; return r; };

  f16* wqkv_h = (f16*)carve(sizeof(f16) * (size_t)LAYERS * Hh * 3 * Hh);
  f16* wo_h   = (f16*)carve(sizeof(f16) * (size_t)LAYERS * Hh * Hh);
  f16* w1_h   = (f16*)carve(sizeof(f16) * (size_t)LAYERS * Hh * FFN);
  f16* w2_h   = (f16*)carve(sizeof(f16) * (size_t)LAYERS * FFN * Hh);
  f16* bpw1_h = (f16*)carve(sizeof(f16) * (size_t)GK * Hh);
  f16* bpw2_h = (f16*)carve(sizeof(f16) * (size_t)Hh * HEADS);
  float* input_emb = (float*)carve(sizeof(float) * (size_t)Bx * Sx * Hh);
  float* out_emb   = (float*)carve(sizeof(float) * (size_t)Bx * Nn * Hh);
  float* em        = (float*)carve(sizeof(float) * 4 * Bx * Nn);
  int*   pmask     = (int*)carve(sizeof(int) * Bx * Sx);
  f16*   gbf_h     = (f16*)carve(sizeof(f16) * (size_t)MBIAS * GK);
  f16*   C1_h      = (f16*)carve(sizeof(f16) * (size_t)MBIAS * Hh);
  float* attn_bias = (float*)carve(sizeof(float) * (size_t)Bx * HEADS * Sx * Sx);
  float* xf32      = (float*)carve(sizeof(float) * (size_t)ROWS * Hh);
  f16*   xh        = (f16*)carve(sizeof(f16) * (size_t)ROWS * Hh);
  float* ybuf      = (float*)carve(sizeof(float) * (size_t)ROWS * Hh);
  f16*   q_h       = (f16*)carve(sizeof(f16) * (size_t)Bx * HEADS * Sx * HD);
  f16*   k_h       = (f16*)carve(sizeof(f16) * (size_t)Bx * HEADS * Sx * HD);
  f16*   v_h       = (f16*)carve(sizeof(f16) * (size_t)Bx * HEADS * Sx * HD);
  f16*   attnout_h = (f16*)carve(sizeof(f16) * (size_t)ROWS * Hh);
  f16*   ffnh_h    = (f16*)carve(sizeof(f16) * (size_t)ROWS * FFN);
  (void)ws_size; (void)n_in; (void)in_sizes; (void)out_size;

  auto cvt = [&](const float* s, f16* d, size_t n) {
    int blocks = (int)((n + 2047) / 2048);
    k_cvt<<<blocks, 256, 0, stream>>>(s, d, (int)n);
  };
  cvt(w_qkv, wqkv_h, (size_t)LAYERS * Hh * 3 * Hh);
  cvt(w_o,   wo_h,   (size_t)LAYERS * Hh * Hh);
  cvt(w1,    w1_h,   (size_t)LAYERS * Hh * FFN);
  cvt(w2,    w2_h,   (size_t)LAYERS * FFN * Hh);
  cvt(bp_w1, bpw1_h, (size_t)GK * Hh);
  cvt(bp_w2, bpw2_h, (size_t)Hh * HEADS);

  k_embed<<<Bx * Nn, 256, 0, stream>>>(input_nodes, atom_in, atom_out, gbf_mul_w,
                                       gbf_bias_w, input_emb, out_emb, em, pmask);
  k_token<<<Bx, 256, 0, stream>>>(graph_tok, input_emb);
  k_ln_emb<<<ROWS, 256, 0, stream>>>(input_emb, emb_ln_g, emb_ln_b, xf32, xh);
  k_gbf<<<MBIAS, GK, 0, stream>>>(pos, em, gbf_means, gbf_stds, gbf_mul_b,
                                  gbf_bias_b, clf_gbf, gbf_h);

  // bias MLP stage 1: [67600,128] @ [128,768] -> gelu -> f16
  {
    GemmP g{gbf_h, bpw1_h, bp_b1, nullptr, nullptr, C1_h, nullptr, nullptr, nullptr,
            MBIAS, Hh, GK};
    k_gemm<EPI_GELU_F16><<<dim3(Hh / 128, (MBIAS + 63) / 64), 256, 0, stream>>>(g);
  }
  k_bias2<<<MBIAS / 8, 256, 0, stream>>>(C1_h, bpw2_h, bp_b2, pmask, attn_bias);

  for (int l = 0; l < LAYERS; ++l) {
    const f16* Wq = wqkv_h + (size_t)l * Hh * 3 * Hh;
    // QKV: [1040,768] @ [768,2304] -> scatter q(scaled)/k/v
    {
      GemmP g{xh, Wq, b_qkv + (size_t)l * 3 * Hh, nullptr, nullptr, nullptr,
              q_h, k_h, v_h, ROWS, 3 * Hh, Hh};
      k_gemm<EPI_QKV><<<dim3(3 * Hh / 128, (ROWS + 63) / 64), 256, 0, stream>>>(g);
    }
    k_attn<<<Bx * HEADS, 256, 0, stream>>>(q_h, k_h, v_h, attn_bias, attnout_h);
    // out-proj + residual
    {
      GemmP g{attnout_h, wo_h + (size_t)l * Hh * Hh, b_o + (size_t)l * Hh,
              xf32, ybuf, nullptr, nullptr, nullptr, nullptr, ROWS, Hh, Hh};
      k_gemm<EPI_F32_RESID><<<dim3(Hh / 128, (ROWS + 63) / 64), 256, 0, stream>>>(g);
    }
    k_ln<<<ROWS, 256, 0, stream>>>(ybuf, ln1_g + (size_t)l * Hh, ln1_b + (size_t)l * Hh,
                                   xf32, xh);
    // FFN1 gelu
    {
      GemmP g{xh, w1_h + (size_t)l * Hh * FFN, b1 + (size_t)l * FFN, nullptr,
              nullptr, ffnh_h, nullptr, nullptr, nullptr, ROWS, FFN, Hh};
      k_gemm<EPI_GELU_F16><<<dim3(FFN / 128, (ROWS + 63) / 64), 256, 0, stream>>>(g);
    }
    // FFN2 + residual
    {
      GemmP g{ffnh_h, w2_h + (size_t)l * FFN * Hh, b2 + (size_t)l * Hh,
              xf32, ybuf, nullptr, nullptr, nullptr, nullptr, ROWS, Hh, FFN};
      k_gemm<EPI_F32_RESID><<<dim3(Hh / 128, (ROWS + 63) / 64), 256, 0, stream>>>(g);
    }
    float* lnout = (l == LAYERS - 1) ? (float*)d_out : xf32;
    k_ln<<<ROWS, 256, 0, stream>>>(ybuf, ln2_g + (size_t)l * Hh, ln2_b + (size_t)l * Hh,
                                   lnout, xh);
  }
}
